// SelfAttentionLSTMEncoder_50319836840020
// MI455X (gfx1250) — matrix operand order repaired and checked by reading, not compiled, not run on hardware
//
#include <hip/hip_runtime.h>
#include <hip/hip_bf16.h>

// ---------------------------------------------------------------------------
// MI455X (gfx1250, wave32, WMMA) implementation of the BiLSTM + self-attention
// NER encoder.
//
//  * ~64 GFLOP total; everything fits in L2 (192MB) -> latency/compute bound.
//  * LSTM recurrences: one persistent 1024-thread workgroup per (layer,dir):
//    per-step x-projection fragments are async-staged into LDS
//    (global_load_async_to_lds_b128 / s_wait_asynccnt), h fragments live in
//    LDS, cell state lives in registers, gate buffer (32x1600 f32) in LDS.
//    Total ~284KB of the 320KB WGP LDS.
//  * All GEMMs: bf16 v_wmma_f32_16x16x32_bf16 with fp32 accumulate over
//    pre-packed per-lane fragments (ISA 16x32 A / 32x16 B layouts).
// ---------------------------------------------------------------------------

typedef __attribute__((ext_vector_type(16))) __bf16 bf16x16;
typedef __attribute__((ext_vector_type(8)))  float  f32x8;

#define DEVINL __device__ __forceinline__

DEVINL f32x8 wmma_bf16(bf16x16 a, bf16x16 b, f32x8 c) {
  // 8 args: (neg_a, A, neg_b, B, c_mod, C, reuse_a, reuse_b)
  return __builtin_amdgcn_wmma_f32_16x16x32_bf16(false, a, false, b, (short)0, c,
                                                 false, false);
}

// CDNA5 async global->LDS copy (16B per lane), tracked by ASYNCcnt.
// ldsOff = low 32 bits of a generic shared pointer = LDS byte offset.
DEVINL void async_copy_b128(unsigned ldsOff, const void* gaddr) {
  asm volatile("global_load_async_to_lds_b128 %0, %1, off"
               :: "v"(ldsOff), "v"(gaddr) : "memory");
}
DEVINL void wait_async0() {
  asm volatile("s_wait_asynccnt 0x0" ::: "memory");
}
DEVINL unsigned lds_off(const void* p) {
  return (unsigned)(unsigned long long)p;
}

// A-matrix (16x32 bf16) lane/element -> k mapping (ISA 7.12.2).
DEVINL int a_k_of(int lane, int j) {
  int v = j >> 1, p = j & 1;
  return ((v < 4) ? 0 : 16) + ((lane >> 4) << 3) + ((v & 3) << 1) + p;
}
// Inverse: element index within a 512-element A tile for (m_local, k_local).
DEVINL int a_pack_idx(int ml, int kl) {
  int half = (kl >> 3) & 1;
  int vv   = ((kl >> 4) << 2) + ((kl & 7) >> 1);
  int p    = kl & 1;
  return ((((half << 4) + ml)) << 4) + (vv << 1) + p;
}
// B-matrix (32x16 bf16): lane holds column n=lane&15, k = (j<8 ? 0 : 16) + (lane>>4)*8 + (j&7).
DEVINL int b_k_of(int lane, int j) { return ((j < 8) ? 0 : 16) + ((lane >> 4) << 3) + (j & 7); }

DEVINL float sigmoidf(float x) { return 1.f / (1.f + expf(-x)); }

// ---------------------------------------------------------------------------
// Generic packing kernels: f32 -> bf16 WMMA fragments.
// A fragments laid out [Mtile][Ktile][512]; B fragments [Ntile][Ktile][512].
// ---------------------------------------------------------------------------
__global__ void pack_A(__bf16* __restrict__ dst, const float* __restrict__ src,
                       long Mtiles, int Ktiles, int K, int ld, int timeMajor) {
  long idx = (long)blockIdx.x * blockDim.x + threadIdx.x;
  long total = Mtiles * (long)Ktiles * 512;
  if (idx >= total) return;
  int  e    = (int)(idx & 511);
  long tIdx = idx >> 9;
  int  kt   = (int)(tIdx % Ktiles);
  long mt   = tIdx / Ktiles;
  int  lane = e >> 4, j = e & 15;
  long m = mt * 16 + (lane & 15);
  int  k = kt * 32 + a_k_of(lane, j);
  // timeMajor: fragment row m = t*32 + b  ->  source row = b*256 + t
  long r = timeMajor ? ((m & 31) * 256 + (m >> 5)) : m;
  float v = (k < K) ? src[r * (long)ld + k] : 0.f;
  dst[idx] = (__bf16)v;
}

__global__ void pack_B(__bf16* __restrict__ dst, const float* __restrict__ src,
                       int Ntiles, int Ktiles, int K, int N, int ld, int trans,
                       long srcBS, long dstBS) {
  long idx = (long)blockIdx.x * blockDim.x + threadIdx.x;
  long perBatch = (long)Ntiles * Ktiles * 512;
  if (idx >= perBatch) return;
  int  b    = blockIdx.y;
  int  e    = (int)(idx & 511);
  long tIdx = idx >> 9;
  int  kt   = (int)(tIdx % Ktiles);
  int  nt   = (int)(tIdx / Ktiles);
  int  lane = e >> 4, j = e & 15;
  int  n = nt * 16 + (lane & 15);
  int  k = kt * 32 + b_k_of(lane, j);
  const float* s = src + (long)b * srcBS;
  float v = 0.f;
  if (k < K && n < N) v = trans ? s[(long)n * ld + k] : s[(long)k * ld + n];
  dst[(long)b * dstBS + idx] = (__bf16)v;
}

// ---------------------------------------------------------------------------
// Generic batched fragment GEMM: C[b] = A[b] @ B[b] (+ bias[n]).
// One 16x16 output tile per wave; inner loop = 32B frag loads + WMMA.
// ---------------------------------------------------------------------------
__global__ void gemm_frag(const __bf16* __restrict__ Af, const __bf16* __restrict__ Bf,
                          float* __restrict__ C, const float* __restrict__ bias,
                          int Mtiles, int Ntiles, int Ktiles,
                          long ldc, int nrows, int ncols,
                          long aBS, long bBS, long cBS) {
  int lane = threadIdx.x & 31;
  int wave = threadIdx.x >> 5;
  int tile = blockIdx.x * (blockDim.x >> 5) + wave;
  if (tile >= Mtiles * Ntiles) return;   // wave-uniform, EXEC stays full
  int mt = tile / Ntiles, nt = tile - mt * Ntiles;
  const __bf16* Ab = Af + (long)blockIdx.y * aBS + (long)mt * Ktiles * 512;
  const __bf16* Bb = Bf + (long)blockIdx.y * bBS + (long)nt * Ktiles * 512;
  f32x8 acc = {};
  for (int kt = 0; kt < Ktiles; ++kt) {
    if (kt + 1 < Ktiles) {
      __builtin_prefetch(Ab + (kt + 1) * 512 + lane * 16, 0, 3);
      __builtin_prefetch(Bb + (kt + 1) * 512 + lane * 16, 0, 3);
    }
    bf16x16 a = *(const bf16x16*)(Ab + kt * 512 + lane * 16);
    bf16x16 b = *(const bf16x16*)(Bb + kt * 512 + lane * 16);
    acc = wmma_bf16(a, b, acc);
  }
  float* Cb = C + (long)blockIdx.y * cBS;
  int n  = nt * 16 + (lane & 15);
  int mb = mt * 16 + ((lane >> 4) << 3);
  float bv = bias ? bias[n] : 0.f;
  union { f32x8 v; float f[8]; } u; u.v = acc;
  if (n < ncols) {
    for (int r = 0; r < 8; ++r) {
      int m = mb + r;
      if (m < nrows) Cb[(long)m * ldc + n] = u.f[r] + bv;
    }
  }
}

// ---------------------------------------------------------------------------
// Gather kernels.
// ---------------------------------------------------------------------------
__global__ void gather_feats(float* __restrict__ feats, const int* __restrict__ words,
                             const int* __restrict__ cases, const int* __restrict__ pos,
                             const float* __restrict__ we, const float* __restrict__ ce,
                             const float* __restrict__ pe) {
  long idx = (long)blockIdx.x * blockDim.x + threadIdx.x;
  if (idx >= (long)8192 * 448) return;
  long r = idx / 448;
  int  k = (int)(idx - r * 448);
  float v = 0.f;
  if (k < 300)      v = we[(long)words[r] * 300 + k];
  else if (k < 400) v = 0.f;                       // chv: filled by char LSTM
  else if (k < 410) v = ce[cases[r] * 10 + (k - 400)];
  else if (k < 440) v = pe[pos[r] * 30 + (k - 410)];
  feats[idx] = v;
}

// Fused char-embedding gather + A-fragment pack.
// Layout: [seqTile(512)][l(16)][512 elems].
__global__ void gather_cheA(__bf16* __restrict__ dst, const int* __restrict__ chars,
                            const float* __restrict__ che) {
  long idx = (long)blockIdx.x * blockDim.x + threadIdx.x;
  if (idx >= (long)512 * 16 * 512) return;
  int  e    = (int)(idx & 511);
  long rest = idx >> 9;
  int  l    = (int)(rest & 15);
  long st   = rest >> 4;
  int  lane = e >> 4, j = e & 15;
  long s = st * 16 + (lane & 15);
  int  k = a_k_of(lane, j);
  float v = (k < 20) ? che[(long)chars[s * 16 + l] * 20 + k] : 0.f;
  dst[idx] = (__bf16)v;
}

// ---------------------------------------------------------------------------
// Elementwise / reduction kernels.
// ---------------------------------------------------------------------------
__global__ void combine_bias(float* dst, const float* a, const float* b, int n, int np) {
  int i = blockIdx.x * blockDim.x + threadIdx.x;
  if (i >= np) return;
  dst[i] = (i < n) ? (a[i] + (b ? b[i] : 0.f)) : 0.f;
}

__global__ void tanh_inplace(float* x, long n) {
  long i = (long)blockIdx.x * blockDim.x + threadIdx.x;
  if (i < n) x[i] = tanhf(x[i]);
}

// One wave32 per (b,t) row of 256 scores: scale 1/sqrt(100), mask, softmax.
__global__ void softmax_rows(float* __restrict__ sc, const int* __restrict__ words) {
  int wid  = (int)((blockIdx.x * (long)blockDim.x + threadIdx.x) >> 5);
  int lane = threadIdx.x & 31;
  if (wid >= 8192) return;
  int b = wid >> 8;
  float* row = sc + (long)wid * 256;
  float v[8]; float mx = -3.0e38f;
  for (int i = 0; i < 8; ++i) {
    int s = lane + (i << 5);
    float x = row[s] * 0.1f;
    if (words[(b << 8) + s] == 0) x = -1e10f;
    v[i] = x; mx = fmaxf(mx, x);
  }
  for (int o = 16; o > 0; o >>= 1) mx = fmaxf(mx, __shfl_xor(mx, o, 32));
  float sum = 0.f;
  for (int i = 0; i < 8; ++i) { v[i] = expf(v[i] - mx); sum += v[i]; }
  for (int o = 16; o > 0; o >>= 1) sum += __shfl_xor(sum, o, 32);
  float inv = 1.f / sum;
  for (int i = 0; i < 8; ++i) row[lane + (i << 5)] = v[i] * inv;
}

// One wave32 per row: layernorm over the first 100 of 112 cols.
__global__ void layer_norm_rows(const float* __restrict__ x, float* __restrict__ y,
                                const float* __restrict__ g, const float* __restrict__ b) {
  int wid  = (int)((blockIdx.x * (long)blockDim.x + threadIdx.x) >> 5);
  int lane = threadIdx.x & 31;
  if (wid >= 8192) return;
  const float* xr = x + (long)wid * 112;
  float s = 0.f, ss = 0.f;
  for (int k = lane; k < 100; k += 32) { float t = xr[k]; s += t; ss += t * t; }
  for (int o = 16; o > 0; o >>= 1) { s += __shfl_xor(s, o, 32); ss += __shfl_xor(ss, o, 32); }
  float mu  = s * 0.01f;
  float var = ss * 0.01f - mu * mu;
  float rs  = rsqrtf(var + 1e-5f);
  float* yr = y + (long)wid * 112;
  for (int k = lane; k < 112; k += 32)
    yr[k] = (k < 100) ? (xr[k] - mu) * rs * g[k] + b[k] : 0.f;
}

// ---------------------------------------------------------------------------
// Char BiLSTM: grid (512 seq-tiles, 2 dirs), 128 threads (4 waves).
// Wih (13x1) + Whh (13x2) fragments async-staged into LDS once; 16 recurrent
// steps, each = 13 tiles x 3 WMMAs + fused gates (cell state in registers);
// final h written into the chv slice of feats.
// ---------------------------------------------------------------------------
__global__ void char_lstm(const __bf16* __restrict__ cheA,
                          const __bf16* __restrict__ WihF, const __bf16* __restrict__ WihBk,
                          const __bf16* __restrict__ WhhF, const __bf16* __restrict__ WhhBk,
                          const float* __restrict__ cbF, const float* __restrict__ cbBk,
                          float* __restrict__ feats) {
  extern __shared__ char smem[];
  __bf16* sWih = (__bf16*)smem;          // 13*512
  __bf16* sWhh = sWih + 13 * 512;        // 26*512
  __bf16* sA   = sWhh + 26 * 512;        // 2*512 (h fragments, 16x64)
  float*  sG   = (float*)(sA + 2 * 512); // 16*208 gate staging
  const int dir = blockIdx.y;
  const long st = blockIdx.x;
  const __bf16* Wih = dir ? WihBk : WihF;
  const __bf16* Whh = dir ? WhhBk : WhhF;
  const float*  cb  = dir ? cbBk : cbF;
  const int lane = threadIdx.x & 31, wave = threadIdx.x >> 5;

  // Async-stage weight fragments into LDS (16B per lane per issue).
  const unsigned oWih = lds_off(sWih), oWhh = lds_off(sWhh);
  for (int i = threadIdx.x; i < (13 * 512) / 8; i += blockDim.x)
    async_copy_b128(oWih + i * 16, Wih + i * 8);
  for (int i = threadIdx.x; i < (26 * 512) / 8; i += blockDim.x)
    async_copy_b128(oWhh + i * 16, Whh + i * 8);
  for (int i = threadIdx.x; i < 2 * 512; i += blockDim.x) sA[i] = (__bf16)0.f;
  float creg[7];
  for (int i = 0; i < 7; ++i) creg[i] = 0.f;
  wait_async0();
  __syncthreads();

  for (int step = 0; step < 16; ++step) {
    int l = dir ? (15 - step) : step;
    const __bf16* ax = cheA + ((st * 16 + l) << 9);
    for (int nt = wave; nt < 13; nt += 4) {
      f32x8 acc = {};
      bf16x16 a  = *(const bf16x16*)(ax + lane * 16);
      bf16x16 bw = *(const bf16x16*)(sWih + nt * 512 + lane * 16);
      acc = wmma_bf16(a, bw, acc);
      for (int kt = 0; kt < 2; ++kt) {
        bf16x16 ah = *(const bf16x16*)(sA + kt * 512 + lane * 16);
        bf16x16 bh = *(const bf16x16*)(sWhh + (nt * 2 + kt) * 512 + lane * 16);
        acc = wmma_bf16(ah, bh, acc);
      }
      int n  = nt * 16 + (lane & 15);
      int mb = (lane >> 4) << 3;
      union { f32x8 v; float f[8]; } u; u.v = acc;
      for (int r = 0; r < 8; ++r) sG[(mb + r) * 208 + n] = u.f[r];
    }
    __syncthreads();
    int it = 0;
    for (int w = threadIdx.x; w < 16 * 50; w += blockDim.x, ++it) {
      int row = w / 50, n = w - row * 50;
      float gi = sG[row * 208 + n]       + cb[n];
      float gf = sG[row * 208 + n + 50]  + cb[n + 50];
      float gg = sG[row * 208 + n + 100] + cb[n + 100];
      float go = sG[row * 208 + n + 150] + cb[n + 150];
      float c = sigmoidf(gf) * creg[it] + sigmoidf(gi) * tanhf(gg);
      float h = sigmoidf(go) * tanhf(c);
      creg[it] = c;
      int kt = n >> 5, kl = n & 31;
      sA[kt * 512 + a_pack_idx(row, kl)] = (__bf16)h;
      if (step == 15)
        feats[(st * 16 + row) * 448 + 300 + dir * 50 + n] = h;
    }
    __syncthreads();
  }
}

// ---------------------------------------------------------------------------
// Main LSTM layer (persistent). grid = 2 (dir), block = 1024 (32 waves).
// LDS: gate buffer 32x1600 f32 (200KB) + staged x-fragments 2*Kx*512 bf16
//      (up to 53KB) + h fragments 2x13x512 bf16 (26KB) -> <=284KB of 320KB.
// Cell state lives in per-thread registers (fixed (row,n) ownership).
// Per step: async-stage x fragments -> 200 output tiles x (Kx+13) WMMAs ->
// fused gates which re-pack h into bf16 fragments for the next step.
// ---------------------------------------------------------------------------
__global__ void lstm_layer(const __bf16* __restrict__ xA, int Kx,
                           const __bf16* __restrict__ WihF, const __bf16* __restrict__ WihBk,
                           const __bf16* __restrict__ WhhF, const __bf16* __restrict__ WhhBk,
                           const float* __restrict__ cbF, const float* __restrict__ cbBk,
                           float* __restrict__ zout) {
  extern __shared__ char smem[];
  float*  sG = (float*)smem;               // 32*1600 f32
  __bf16* sX = (__bf16*)(sG + 32 * 1600);  // 2*Kx*512 bf16 (x fragments)
  __bf16* sA = sX + 2 * Kx * 512;          // 2*13*512 bf16 (h fragments)
  const int dir = blockIdx.x;
  const __bf16* Wih = dir ? WihBk : WihF;
  const __bf16* Whh = dir ? WhhBk : WhhF;
  const float*  cb  = dir ? cbBk : cbF;
  const int lane = threadIdx.x & 31, wave = threadIdx.x >> 5;
  const unsigned oX = lds_off(sX);
  const int xb128 = (2 * Kx * 512) / 8;    // 16B packets per step

  for (int i = threadIdx.x; i < 2 * 13 * 512; i += blockDim.x) sA[i] = (__bf16)0.f;
  float creg[13];
  for (int i = 0; i < 13; ++i) creg[i] = 0.f;
  __syncthreads();

  for (int tt = 0; tt < 256; ++tt) {
    int t = dir ? (255 - tt) : tt;
    // Async-stage this step's x-projection fragments (read 100x below).
    {
      const __bf16* gx = xA + ((long)t * 2 * Kx) * 512;
      for (int i = threadIdx.x; i < xb128; i += blockDim.x)
        async_copy_b128(oX + i * 16, gx + i * 8);
      wait_async0();
      __syncthreads();
    }
    for (int tile = wave; tile < 200; tile += 32) {
      int mt = tile / 100, nt = tile - mt * 100;
      f32x8 acc = {};
      const __bf16* ax = sX + (long)mt * Kx * 512;
      const __bf16* bx = Wih + (long)nt * Kx * 512;
      for (int kt = 0; kt < Kx; ++kt) {
        if (kt + 1 < Kx) __builtin_prefetch(bx + (kt + 1) * 512 + lane * 16, 0, 3);
        bf16x16 a = *(const bf16x16*)(ax + kt * 512 + lane * 16);
        bf16x16 b = *(const bf16x16*)(bx + kt * 512 + lane * 16);
        acc = wmma_bf16(a, b, acc);
      }
      const __bf16* ah = sA + mt * 13 * 512;
      const __bf16* bh = Whh + (long)nt * 13 * 512;
      for (int kt = 0; kt < 13; ++kt) {
        bf16x16 a = *(const bf16x16*)(ah + kt * 512 + lane * 16);
        bf16x16 b = *(const bf16x16*)(bh + kt * 512 + lane * 16);
        acc = wmma_bf16(a, b, acc);
      }
      int n  = nt * 16 + (lane & 15);
      int mb = mt * 16 + ((lane >> 4) << 3);
      union { f32x8 v; float f[8]; } u; u.v = acc;
      for (int r = 0; r < 8; ++r) sG[(mb + r) * 1600 + n] = u.f[r];
    }
    __syncthreads();
    int it = 0;
    for (int w = threadIdx.x; w < 32 * 400; w += blockDim.x, ++it) {
      int row = w / 400, n = w - row * 400;
      float gi = sG[row * 1600 + n]        + cb[n];
      float gf = sG[row * 1600 + n + 400]  + cb[n + 400];
      float gg = sG[row * 1600 + n + 800]  + cb[n + 800];
      float go = sG[row * 1600 + n + 1200] + cb[n + 1200];
      float c = sigmoidf(gf) * creg[it] + sigmoidf(gi) * tanhf(gg);
      float h = sigmoidf(go) * tanhf(c);
      creg[it] = c;
      int mt = row >> 4, ml = row & 15;
      int kt = n >> 5,  kl = n & 31;
      sA[(mt * 13 + kt) * 512 + a_pack_idx(ml, kl)] = (__bf16)h;
      zout[((long)row * 256 + t) * 800 + dir * 400 + n] = h;
    }
    __syncthreads();
  }
}

// ---------------------------------------------------------------------------
// Host orchestration.
// ---------------------------------------------------------------------------
static inline unsigned cdiv(long a, long b) { return (unsigned)((a + b - 1) / b); }

extern "C" void kernel_launch(void* const* d_in, const int* in_sizes, int n_in,
                              void* d_out, int out_size, void* d_ws, size_t ws_size,
                              hipStream_t stream) {
  (void)in_sizes; (void)n_in; (void)out_size; (void)ws_size;
  const int*   words    = (const int*)d_in[0];
  const int*   chars    = (const int*)d_in[1];
  const int*   cases_   = (const int*)d_in[2];
  const int*   pos      = (const int*)d_in[3];
  const float* word_emb = (const float*)d_in[4];
  const float* char_emb = (const float*)d_in[5];
  const float* case_emb = (const float*)d_in[6];
  const float* pos_emb  = (const float*)d_in[7];
  // char_lstm params 8..15, lstm0 16..23, lstm1 24..31 (Wih_f,Whh_f,bih_f,bhh_f, then _b)
  const float* cWih_f = (const float*)d_in[8];  const float* cWhh_f = (const float*)d_in[9];
  const float* cbih_f = (const float*)d_in[10]; const float* cbhh_f = (const float*)d_in[11];
  const float* cWih_b = (const float*)d_in[12]; const float* cWhh_b = (const float*)d_in[13];
  const float* cbih_b = (const float*)d_in[14]; const float* cbhh_b = (const float*)d_in[15];
  const float* W0ih_f = (const float*)d_in[16]; const float* W0hh_f = (const float*)d_in[17];
  const float* b0ih_f = (const float*)d_in[18]; const float* b0hh_f = (const float*)d_in[19];
  const float* W0ih_b = (const float*)d_in[20]; const float* W0hh_b = (const float*)d_in[21];
  const float* b0ih_b = (const float*)d_in[22]; const float* b0hh_b = (const float*)d_in[23];
  const float* W1ih_f = (const float*)d_in[24]; const float* W1hh_f = (const float*)d_in[25];
  const float* b1ih_f = (const float*)d_in[26]; const float* b1hh_f = (const float*)d_in[27];
  const float* W1ih_b = (const float*)d_in[28]; const float* W1hh_b = (const float*)d_in[29];
  const float* b1ih_b = (const float*)d_in[30]; const float* b1hh_b = (const float*)d_in[31];
  const float* ffW = (const float*)d_in[32]; const float* ffb = (const float*)d_in[33];
  const float* QW  = (const float*)d_in[34]; const float* Qb  = (const float*)d_in[35];
  const float* KW  = (const float*)d_in[36]; const float* Kb  = (const float*)d_in[37];
  const float* VW  = (const float*)d_in[38]; const float* Vb  = (const float*)d_in[39];
  const float* lng = (const float*)d_in[40]; const float* lnb = (const float*)d_in[41];
  const float* outW = (const float*)d_in[42]; const float* outb = (const float*)d_in[43];
  float* out = (float*)d_out;

  char*  base = (char*)d_ws;
  size_t off  = 0;
  auto alloc = [&](size_t bytes) -> void* {
    void* p = base + off;
    off = (off + bytes + 1023) & ~(size_t)1023;
    return p;
  };
  auto bfrag = [](long nt, long kt) -> size_t { return (size_t)nt * kt * 512 * 2; };

  // --- workspace layout --------------------------------------------------
  float* cbC_f = (float*)alloc(208 * 4);  float* cbC_b = (float*)alloc(208 * 4);
  float* cb0_f = (float*)alloc(1600 * 4); float* cb0_b = (float*)alloc(1600 * 4);
  float* cb1_f = (float*)alloc(1600 * 4); float* cb1_b = (float*)alloc(1600 * 4);
  float* ffb_p = (float*)alloc(112 * 4);
  float* qb_p  = (float*)alloc(112 * 4);  float* kb_p = (float*)alloc(112 * 4);
  float* vb_p  = (float*)alloc(112 * 4);  float* ob_p = (float*)alloc(32 * 4);

  __bf16* cWihF_B = (__bf16*)alloc(bfrag(13, 1));  __bf16* cWihB_B = (__bf16*)alloc(bfrag(13, 1));
  __bf16* cWhhF_B = (__bf16*)alloc(bfrag(13, 2));  __bf16* cWhhB_B = (__bf16*)alloc(bfrag(13, 2));
  __bf16* W0ihF_B = (__bf16*)alloc(bfrag(100, 14)); __bf16* W0ihB_B = (__bf16*)alloc(bfrag(100, 14));
  __bf16* W0hhF_B = (__bf16*)alloc(bfrag(100, 13)); __bf16* W0hhB_B = (__bf16*)alloc(bfrag(100, 13));
  __bf16* W1ihF_B = (__bf16*)alloc(bfrag(100, 26)); __bf16* W1ihB_B = (__bf16*)alloc(bfrag(100, 26));
  __bf16* W1hhF_B = (__bf16*)alloc(bfrag(100, 13)); __bf16* W1hhB_B = (__bf16*)alloc(bfrag(100, 13));
  __bf16* ffW_B  = (__bf16*)alloc(bfrag(7, 26));
  __bf16* QW_B   = (__bf16*)alloc(bfrag(7, 4));  __bf16* KW_B = (__bf16*)alloc(bfrag(7, 4));
  __bf16* VW_B   = (__bf16*)alloc(bfrag(7, 4));  __bf16* outW_B = (__bf16*)alloc(bfrag(2, 4));

  __bf16* cheA   = (__bf16*)alloc((size_t)512 * 16 * 512 * 2);
  float*  feats  = (float*)alloc((size_t)8192 * 448 * 4);
  __bf16* featsA = (__bf16*)alloc((size_t)512 * 14 * 512 * 2);
  float*  z0     = (float*)alloc((size_t)8192 * 800 * 4);
  __bf16* z0A    = (__bf16*)alloc((size_t)512 * 26 * 512 * 2);
  float*  z1     = (float*)alloc((size_t)8192 * 800 * 4);
  __bf16* z1A    = (__bf16*)alloc((size_t)512 * 26 * 512 * 2);
  float*  zf     = (float*)alloc((size_t)8192 * 112 * 4);
  __bf16* zfA    = (__bf16*)alloc((size_t)512 * 4 * 512 * 2);
  float*  qbuf   = (float*)alloc((size_t)8192 * 112 * 4);
  float*  kbuf   = (float*)alloc((size_t)8192 * 112 * 4);
  float*  vbuf   = (float*)alloc((size_t)8192 * 112 * 4);
  __bf16* qA     = (__bf16*)alloc((size_t)512 * 4 * 512 * 2);
  __bf16* kB     = (__bf16*)alloc((size_t)32 * 16 * 4 * 512 * 2);
  float*  scores = (float*)alloc((size_t)8192 * 256 * 4);
  __bf16* attnA  = (__bf16*)alloc((size_t)512 * 8 * 512 * 2);
  __bf16* vB     = (__bf16*)alloc((size_t)32 * 7 * 8 * 512 * 2);
  float*  zatt   = (float*)alloc((size_t)8192 * 112 * 4);
  float*  zln    = (float*)alloc((size_t)8192 * 112 * 4);
  __bf16* zlnA   = (__bf16*)alloc((size_t)512 * 4 * 512 * 2);

  const int TPB = 256;

  // --- combined / padded biases -----------------------------------------
  combine_bias<<<cdiv(208, TPB), TPB, 0, stream>>>(cbC_f, cbih_f, cbhh_f, 200, 208);
  combine_bias<<<cdiv(208, TPB), TPB, 0, stream>>>(cbC_b, cbih_b, cbhh_b, 200, 208);
  combine_bias<<<cdiv(1600, TPB), TPB, 0, stream>>>(cb0_f, b0ih_f, b0hh_f, 1600, 1600);
  combine_bias<<<cdiv(1600, TPB), TPB, 0, stream>>>(cb0_b, b0ih_b, b0hh_b, 1600, 1600);
  combine_bias<<<cdiv(1600, TPB), TPB, 0, stream>>>(cb1_f, b1ih_f, b1hh_f, 1600, 1600);
  combine_bias<<<cdiv(1600, TPB), TPB, 0, stream>>>(cb1_b, b1ih_b, b1hh_b, 1600, 1600);
  combine_bias<<<cdiv(112, TPB), TPB, 0, stream>>>(ffb_p, ffb, nullptr, 100, 112);
  combine_bias<<<cdiv(112, TPB), TPB, 0, stream>>>(qb_p, Qb, nullptr, 100, 112);
  combine_bias<<<cdiv(112, TPB), TPB, 0, stream>>>(kb_p, Kb, nullptr, 100, 112);
  combine_bias<<<cdiv(112, TPB), TPB, 0, stream>>>(vb_p, Vb, nullptr, 100, 112);
  combine_bias<<<cdiv(32, TPB), TPB, 0, stream>>>(ob_p, outb, nullptr, 20, 32);

  // --- weight fragment packing (B = W^T) --------------------------------
  auto packW = [&](__bf16* dst, const float* src, int Nt, int Kt, int K, int N, int ld) {
    long n = (long)Nt * Kt * 512;
    pack_B<<<dim3(cdiv(n, TPB), 1), TPB, 0, stream>>>(dst, src, Nt, Kt, K, N, ld, 1, 0, 0);
  };
  packW(cWihF_B, cWih_f, 13, 1, 20, 200, 20);   packW(cWihB_B, cWih_b, 13, 1, 20, 200, 20);
  packW(cWhhF_B, cWhh_f, 13, 2, 50, 200, 50);   packW(cWhhB_B, cWhh_b, 13, 2, 50, 200, 50);
  packW(W0ihF_B, W0ih_f, 100, 14, 440, 1600, 440); packW(W0ihB_B, W0ih_b, 100, 14, 440, 1600, 440);
  packW(W0hhF_B, W0hh_f, 100, 13, 400, 1600, 400); packW(W0hhB_B, W0hh_b, 100, 13, 400, 1600, 400);
  packW(W1ihF_B, W1ih_f, 100, 26, 800, 1600, 800); packW(W1ihB_B, W1ih_b, 100, 26, 800, 1600, 800);
  packW(W1hhF_B, W1hh_f, 100, 13, 400, 1600, 400); packW(W1hhB_B, W1hh_b, 100, 13, 400, 1600, 400);
  packW(ffW_B, ffW, 7, 26, 800, 100, 800);
  packW(QW_B, QW, 7, 4, 100, 100, 100);
  packW(KW_B, KW, 7, 4, 100, 100, 100);
  packW(VW_B, VW, 7, 4, 100, 100, 100);
  packW(outW_B, outW, 2, 4, 100, 20, 100);

  // --- embeddings --------------------------------------------------------
  gather_feats<<<cdiv((long)8192 * 448, TPB), TPB, 0, stream>>>(
      feats, words, cases_, pos, word_emb, case_emb, pos_emb);
  gather_cheA<<<cdiv((long)512 * 16 * 512, TPB), TPB, 0, stream>>>(cheA, chars, char_emb);

  // --- char BiLSTM (writes chv slice of feats) ---------------------------
  const unsigned CHAR_SMEM = (13 * 512 + 26 * 512 + 2 * 512) * 2 + (16 * 208) * 4;
  char_lstm<<<dim3(512, 2), 128, CHAR_SMEM, stream>>>(
      cheA, cWihF_B, cWihB_B, cWhhF_B, cWhhB_B, cbC_f, cbC_b, feats);

  // --- layer 0 -----------------------------------------------------------
  pack_A<<<cdiv((long)512 * 14 * 512, TPB), TPB, 0, stream>>>(
      featsA, feats, 512, 14, 440, 448, /*timeMajor=*/1);
  auto lstm_smem = [](int Kx) -> unsigned {
    return (32 * 1600) * 4 + (2 * Kx * 512 + 2 * 13 * 512) * 2;
  };
  lstm_layer<<<2, 1024, lstm_smem(14), stream>>>(
      featsA, 14, W0ihF_B, W0ihB_B, W0hhF_B, W0hhB_B, cb0_f, cb0_b, z0);

  // --- layer 1 -----------------------------------------------------------
  pack_A<<<cdiv((long)512 * 26 * 512, TPB), TPB, 0, stream>>>(
      z0A, z0, 512, 26, 800, 800, /*timeMajor=*/1);
  lstm_layer<<<2, 1024, lstm_smem(26), stream>>>(
      z0A, 26, W1ihF_B, W1ihB_B, W1hhF_B, W1hhB_B, cb1_f, cb1_b, z1);

  // --- FF + tanh ---------------------------------------------------------
  pack_A<<<cdiv((long)512 * 26 * 512, TPB), TPB, 0, stream>>>(z1A, z1, 512, 26, 800, 800, 0);
  gemm_frag<<<dim3(cdiv(512L * 7, 8), 1), 256, 0, stream>>>(
      z1A, ffW_B, zf, ffb_p, 512, 7, 26, 112, 8192, 112, 0, 0, 0);
  tanh_inplace<<<cdiv((long)8192 * 112, TPB), TPB, 0, stream>>>(zf, (long)8192 * 112);

  // --- Q/K/V projections -------------------------------------------------
  pack_A<<<cdiv((long)512 * 4 * 512, TPB), TPB, 0, stream>>>(zfA, zf, 512, 4, 112, 112, 0);
  gemm_frag<<<dim3(cdiv(512L * 7, 8), 1), 256, 0, stream>>>(
      zfA, QW_B, qbuf, qb_p, 512, 7, 4, 112, 8192, 112, 0, 0, 0);
  gemm_frag<<<dim3(cdiv(512L * 7, 8), 1), 256, 0, stream>>>(
      zfA, KW_B, kbuf, kb_p, 512, 7, 4, 112, 8192, 112, 0, 0, 0);
  gemm_frag<<<dim3(cdiv(512L * 7, 8), 1), 256, 0, stream>>>(
      zfA, VW_B, vbuf, vb_p, 512, 7, 4, 112, 8192, 112, 0, 0, 0);

  // --- attention scores: per batch Q[256x128] x K^T[128x256] -------------
  pack_A<<<cdiv((long)512 * 4 * 512, TPB), TPB, 0, stream>>>(qA, qbuf, 512, 4, 112, 112, 0);
  pack_B<<<dim3(cdiv((long)16 * 4 * 512, TPB), 32), TPB, 0, stream>>>(
      kB, kbuf, 16, 4, 112, 256, 112, /*trans=*/1, 256L * 112, 16L * 4 * 512);
  gemm_frag<<<dim3(cdiv(16L * 16, 8), 32), 256, 0, stream>>>(
      qA, kB, scores, nullptr, 16, 16, 4, 256, 256, 256,
      16L * 4 * 512, 16L * 4 * 512, 256L * 256);

  // --- softmax (scale + mask fused) -------------------------------------
  softmax_rows<<<cdiv(8192L * 32, TPB), TPB, 0, stream>>>(scores, words);

  // --- attn @ V ----------------------------------------------------------
  pack_A<<<cdiv((long)512 * 8 * 512, TPB), TPB, 0, stream>>>(attnA, scores, 512, 8, 256, 256, 0);
  pack_B<<<dim3(cdiv((long)7 * 8 * 512, TPB), 32), TPB, 0, stream>>>(
      vB, vbuf, 7, 8, 256, 112, 112, /*trans=*/0, 256L * 112, 7L * 8 * 512);
  gemm_frag<<<dim3(cdiv(16L * 7, 8), 32), 256, 0, stream>>>(
      attnA, vB, zatt, nullptr, 16, 7, 8, 112, 256, 112,
      16L * 8 * 512, 7L * 8 * 512, 256L * 112);

  // --- layernorm + output projection -------------------------------------
  layer_norm_rows<<<cdiv(8192L * 32, TPB), TPB, 0, stream>>>(zatt, zln, lng, lnb);
  pack_A<<<cdiv((long)512 * 4 * 512, TPB), TPB, 0, stream>>>(zlnA, zln, 512, 4, 112, 112, 0);
  gemm_frag<<<dim3(cdiv(512L * 2, 8), 1), 256, 0, stream>>>(
      zlnA, outW_B, out, ob_p, 512, 2, 4, 20, 8192, 20, 0, 0, 0);
}
